// MultiHead_8624294330576
// MI455X (gfx1250) — matrix operand order repaired and checked by reading, not compiled, not run on hardware
//
#include <hip/hip_runtime.h>
#include <hip/hip_bf16.h>

// ---------------------------------------------------------------------------
// MultiHead attention for MI455X (gfx1250, wave32, WMMA + async LDS staging).
//   B=4, N=2048, EMBED=128, H=8, HEAD_DIM=16, VDIM=16, SCALING=0.25
// QK^T in exact fp32 (v_wmma_f32_16x16x4_f32, softmax is sharp). P*V via
// v_wmma_f32_16x16x32_f16. K/V tiles are staged once per block into
// double-buffered LDS with global_load_async_to_lds_b128 (ASYNCcnt),
// removing the 8x redundant L2 reads across the 8 waves of a block.
// ---------------------------------------------------------------------------

typedef float    v2f  __attribute__((ext_vector_type(2)));
typedef float    v8f  __attribute__((ext_vector_type(8)));
typedef _Float16 v16h __attribute__((ext_vector_type(16)));

#define BB   4
#define NN   2048
#define EMB  128
#define HH   8
#define HD   16

static __device__ __forceinline__ v8f wmma_f32x4(v2f a, v2f b, v8f c) {
  return __builtin_amdgcn_wmma_f32_16x16x4_f32(false, a, false, b, (short)0, c,
                                               false, false);
}
static __device__ __forceinline__ v8f wmma_f16x32(v16h a, v16h b, v8f c) {
  return __builtin_amdgcn_wmma_f32_16x16x32_f16(false, a, false, b, (short)0, c,
                                                false, false);
}

// async global -> LDS copy of 16 bytes per lane (ASYNCcnt-tracked DMA)
static __device__ __forceinline__ void async_ld16(unsigned lds_off,
                                                  const void* gptr) {
  unsigned long long ga = (unsigned long long)gptr;
  asm volatile("global_load_async_to_lds_b128 %0, %1, off"
               :: "v"(lds_off), "v"(ga)
               : "memory");
}
static __device__ __forceinline__ void wait_async0() {
  asm volatile("s_wait_asynccnt 0x0" ::: "memory");
}
static __device__ __forceinline__ unsigned lds_addr(const void* p) {
  return (unsigned)(unsigned long long)p;  // flat LDS aperture: addr[31:0]
}

// ---------------------------------------------------------------------------
// Y[M,128] = (X[M,128] @ W[128,128]^T + bias) * scale     (fp32 WMMA)
// ---------------------------------------------------------------------------
__global__ void proj128_kernel(const float* __restrict__ X,
                               const float* __restrict__ W,
                               const float* __restrict__ bias, float scale,
                               float* __restrict__ Y) {
  const int lane = threadIdx.x & 31;
  const int wave = threadIdx.x >> 5;
  const int tile = blockIdx.x * 8 + wave;  // 512 row-tiles * 8 col-tiles
  const int trow = tile >> 3;
  const int tcol = tile & 7;
  const int m = lane & 15;
  const int g = lane >> 4;

  const float* xrow = X + (size_t)(trow * 16 + m) * 128;
  const float* wrow = W + (size_t)(tcol * 16 + m) * 128;

  v8f acc = {};
#pragma unroll
  for (int k0 = 0; k0 < 128; k0 += 4) {
    v2f a = *(const v2f*)(xrow + k0 + 2 * g);
    v2f b = *(const v2f*)(wrow + k0 + 2 * g);
    acc = wmma_f32x4(a, b, acc);
  }
  const float bn = bias[tcol * 16 + m];
#pragma unroll
  for (int r = 0; r < 8; ++r) {
    const int row = trow * 16 + r + 8 * g;
    Y[(size_t)row * 128 + tcol * 16 + m] = (acc[r] + bn) * scale;
  }
}

// ---------------------------------------------------------------------------
// V projection: X[M,16] @ Wv[128,16]^T + bv, stored f16 TRANSPOSED per head:
//   VT[((b*8 + h)*16 + d) * N + key]
// ---------------------------------------------------------------------------
__global__ void projv_kernel(const float* __restrict__ X,
                             const float* __restrict__ W,
                             const float* __restrict__ bias,
                             _Float16* __restrict__ VT) {
  const int lane = threadIdx.x & 31;
  const int wave = threadIdx.x >> 5;
  const int tile = blockIdx.x * 8 + wave;
  const int trow = tile >> 3;
  const int tcol = tile & 7;  // head index
  const int m = lane & 15;
  const int g = lane >> 4;

  const float* xrow = X + (size_t)(trow * 16 + m) * 16;
  const float* wrow = W + (size_t)(tcol * 16 + m) * 16;

  v8f acc = {};
#pragma unroll
  for (int k0 = 0; k0 < 16; k0 += 4) {
    v2f a = *(const v2f*)(xrow + k0 + 2 * g);
    v2f b = *(const v2f*)(wrow + k0 + 2 * g);
    acc = wmma_f32x4(a, b, acc);
  }
  const float bn = bias[tcol * 16 + m];
#pragma unroll
  for (int r = 0; r < 8; ++r) {
    const int row = trow * 16 + r + 8 * g;  // global row = b*N + key
    const int b   = row >> 11;
    const int key = row & 2047;
    VT[((size_t)((b * 8 + tcol) * 16 + m)) * NN + key] =
        (_Float16)(acc[r] + bn);
  }
}

// ---------------------------------------------------------------------------
// Flash attention. One wave per (b, h, 16-query block); the 8 waves of a
// block share (b,h), so K/V tiles are async-DMA'd into double-buffered LDS
// once per block and consumed by all waves.
// ---------------------------------------------------------------------------
__global__ void attn_kernel(const float* __restrict__ QP,
                            const float* __restrict__ KP,
                            const _Float16* __restrict__ VT,
                            float* __restrict__ CTX) {
  __shared__ __attribute__((aligned(32))) float    kt[2][32 * 16];   // 4 KB
  __shared__ __attribute__((aligned(32))) _Float16 vts[2][16 * 32];  // 2 KB
  __shared__ __attribute__((aligned(32))) _Float16 pbuf_all[8 * 16 * 32]; // 8KB

  const int lane = threadIdx.x & 31;
  const int wave = threadIdx.x >> 5;
  _Float16* pbuf = pbuf_all + wave * (16 * 32);

  const int wg   = blockIdx.x * 8 + wave;  // 4096 waves total
  const int qblk = wg & 127;
  const int h    = (wg >> 7) & 7;  // identical for all 8 waves of a block
  const int b    = wg >> 10;       // identical for all 8 waves of a block
  const int m = lane & 15;
  const int g = lane >> 4;

  // Q A-fragments (fp32, K=16 -> 4 fragments), loaded once, reused 64x.
  const float* qbase =
      QP + ((size_t)(b * NN + qblk * 16 + m)) * 128 + h * 16;
  v2f aq[4];
#pragma unroll
  for (int t = 0; t < 4; ++t) aq[t] = *(const v2f*)(qbase + 4 * t + 2 * g);

  const float*    kbase = KP + ((size_t)b * NN) * 128 + h * 16;
  const _Float16* vbase = VT + ((size_t)((b * 8 + h) * 16)) * NN;

  const unsigned ktA[2]  = {lds_addr(&kt[0][0]), lds_addr(&kt[1][0])};
  const unsigned vtsA[2] = {lds_addr(&vts[0][0]), lds_addr(&vts[1][0])};

  // Issue async copies for one 32-key tile (wave 0 only):
  //   K tile: 32 keys x 16 dims fp32 = 2 KB  -> 4 x b128/lane
  //   V tile: 16 dims x 32 keys f16  = 1 KB  -> 2 x b128/lane
  auto issue_tile = [&](int bf, int key0) {
#pragma unroll
    for (int i = 0; i < 4; ++i) {
      const int c = i * 32 + lane;          // 16-byte chunk id, 0..127
      const int key = c >> 2, part = c & 3;
      async_ld16(ktA[bf] + (unsigned)(key * 64 + part * 16),
                 kbase + (size_t)(key0 + key) * 128 + part * 4);
    }
#pragma unroll
    for (int i = 0; i < 2; ++i) {
      const int c = i * 32 + lane;          // 16-byte chunk id, 0..63
      const int d = c >> 2, part = c & 3;
      async_ld16(vtsA[bf] + (unsigned)(d * 64 + part * 16),
                 vbase + (size_t)d * NN + key0 + part * 8);
    }
  };

  float rmax[8], rsum[8];
#pragma unroll
  for (int r = 0; r < 8; ++r) { rmax[r] = -1e30f; rsum[r] = 0.0f; }
  v8f oacc = {};

  if (wave == 0) { issue_tile(0, 0); wait_async0(); }
  __syncthreads();

  int buf = 0;
  for (int key0 = 0; key0 < NN; key0 += 32) {
    if (wave == 0 && key0 + 32 < NN) issue_tile(buf ^ 1, key0 + 32);

    // scores for 32 keys from the LDS K tile (two fp32 WMMA tiles)
    v8f sc0 = {}, sc1 = {};
#pragma unroll
    for (int t = 0; t < 4; ++t) {
      v2f bk = *(const v2f*)(&kt[buf][m * 16 + 4 * t + 2 * g]);
      sc0 = wmma_f32x4(aq[t], bk, sc0);
    }
#pragma unroll
    for (int t = 0; t < 4; ++t) {
      v2f bk = *(const v2f*)(&kt[buf][(16 + m) * 16 + 4 * t + 2 * g]);
      sc1 = wmma_f32x4(aq[t], bk, sc1);
    }

    // online softmax; row r+8g lives across the 16 lanes of this half-wave
#pragma unroll
    for (int r = 0; r < 8; ++r) {
      float tmax = fmaxf(sc0[r], sc1[r]);
#pragma unroll
      for (int msk = 8; msk >= 1; msk >>= 1)
        tmax = fmaxf(tmax, __shfl_xor(tmax, msk, 32));
      const float nm    = fmaxf(rmax[r], tmax);
      const float alpha = __expf(rmax[r] - nm);
      const float p0 = __expf(sc0[r] - nm);
      const float p1 = __expf(sc1[r] - nm);
      float ts = p0 + p1;
#pragma unroll
      for (int msk = 8; msk >= 1; msk >>= 1) ts += __shfl_xor(ts, msk, 32);
      rsum[r] = rsum[r] * alpha + ts;
      rmax[r] = nm;
      oacc[r] *= alpha;
      const int row = r + 8 * g;
      pbuf[row * 32 + m]      = (_Float16)p0;  // keys key0 + m
      pbuf[row * 32 + 16 + m] = (_Float16)p1;  // keys key0 + 16 + m
    }

    // Reload P in f16 A-fragment layout (wave-internal LDS, in-order DS ops)
    v16h pa;
#pragma unroll
    for (int v = 0; v < 8; ++v) {
      const int kb = 16 * g + 2 * v;
      pa[2 * v]     = pbuf[m * 32 + kb];
      pa[2 * v + 1] = pbuf[m * 32 + kb + 1];
    }
    // V B-fragment from LDS: lane holds column d=m; halves keys 16g..16g+15
    v16h vb = *(const v16h*)(&vts[buf][m * 32 + 16 * g]);
    oacc = wmma_f16x32(pa, vb, oacc);

    if (wave == 0) wait_async0();  // next tile landed
    __syncthreads();               // + everyone done reading this tile
    buf ^= 1;
  }

  float* crow = CTX + ((size_t)(b * NN + qblk * 16)) * 128 + h * 16 + m;
#pragma unroll
  for (int r = 0; r < 8; ++r) {
    const int row = r + 8 * g;
    crow[(size_t)row * 128] = oacc[r] / rsum[r];
  }
}

// ---------------------------------------------------------------------------
// out[M,16] = CTX[M,128] @ Wo[16,128]^T + bo   (fp32 WMMA)
// ---------------------------------------------------------------------------
__global__ void outproj_kernel(const float* __restrict__ CTX,
                               const float* __restrict__ Wo,
                               const float* __restrict__ bo,
                               float* __restrict__ OUT) {
  const int lane = threadIdx.x & 31;
  const int wave = threadIdx.x >> 5;
  const int trow = blockIdx.x * 8 + wave;  // 512 row tiles
  const int m = lane & 15;
  const int g = lane >> 4;

  const float* xrow = CTX + (size_t)(trow * 16 + m) * 128;
  const float* wrow = Wo + (size_t)m * 128;

  v8f acc = {};
#pragma unroll
  for (int k0 = 0; k0 < 128; k0 += 4) {
    v2f a = *(const v2f*)(xrow + k0 + 2 * g);
    v2f b = *(const v2f*)(wrow + k0 + 2 * g);
    acc = wmma_f32x4(a, b, acc);
  }
  const float bn = bo[m];
#pragma unroll
  for (int r = 0; r < 8; ++r)
    OUT[(size_t)(trow * 16 + r + 8 * g) * 16 + m] = acc[r] + bn;
}

// ---------------------------------------------------------------------------
extern "C" void kernel_launch(void* const* d_in, const int* in_sizes, int n_in,
                              void* d_out, int out_size, void* d_ws,
                              size_t ws_size, hipStream_t stream) {
  const float* query = (const float*)d_in[0];
  const float* key   = (const float*)d_in[1];
  const float* value = (const float*)d_in[2];
  const float* Wq    = (const float*)d_in[3];
  const float* bq    = (const float*)d_in[4];
  const float* Wk    = (const float*)d_in[5];
  const float* bk    = (const float*)d_in[6];
  const float* Wv    = (const float*)d_in[7];
  const float* bv    = (const float*)d_in[8];
  const float* Wo    = (const float*)d_in[9];
  const float* bo    = (const float*)d_in[10];

  // Workspace layout (14 MB): qp 4MB | kp 4MB | vt 2MB | ctx 4MB
  char* ws = (char*)d_ws;
  float*    qp  = (float*)(ws);
  float*    kp  = (float*)(ws + (size_t)4 * 1024 * 1024);
  _Float16* vt  = (_Float16*)(ws + (size_t)8 * 1024 * 1024);
  float*    ctx = (float*)(ws + (size_t)10 * 1024 * 1024);
  float*    out = (float*)d_out;

  proj128_kernel<<<512, 256, 0, stream>>>(query, Wq, bq, 0.25f, qp);
  proj128_kernel<<<512, 256, 0, stream>>>(key, Wk, bk, 1.0f, kp);
  projv_kernel<<<512, 256, 0, stream>>>(value, Wv, bv, vt);
  attn_kernel<<<512, 256, 0, stream>>>(qp, kp, vt, ctx);
  outproj_kernel<<<64, 256, 0, stream>>>(ctx, Wo, bo, out);
}
